// cseft_66709432042166
// MI455X (gfx1250) — matrix operand order repaired and checked by reading, not compiled, not run on hardware
//
#include <hip/hip_runtime.h>
#include <hip/hip_bf16.h>

// ---------------------------------------------------------------------------
// Set-transformer cross-attention, MI455X / gfx1250, wave32 WMMA.
//   cvt(x,y)->f16 ; transpose(W*)->f16 ; 3x projection GEMM (WMMA f16) with
//   head-major q/k and transposed v ; fused attention with LDS-staged K/V
//   tiles (GLOBAL_LOAD_ASYNC_TO_LDS_B128, ASYNCcnt) shared by 8 waves/block.
// L2 traffic for attention drops 8x vs per-wave fetch -> WMMA-bound.
// Workspace: 35 MB.
// ---------------------------------------------------------------------------

typedef _Float16 v16h __attribute__((ext_vector_type(16)));
typedef _Float16 v8h  __attribute__((ext_vector_type(8)));
typedef float    v8f  __attribute__((ext_vector_type(8)));

#define N_ROWS 4096
#define D_IN   512
#define NHEAD  8
#define DH     128
#define HD     1024   // NHEAD*DH
#define NITEM  256
#define NSET   16     // N_ROWS / NITEM
#define MTW    8      // mt tiles (= waves) per attention block

#define SHUF16(a, b) __builtin_shufflevector((a), (b), 0,1,2,3,4,5,6,7,8,9,10,11,12,13,14,15)

#define WMMA_F16(A, B, C) \
    __builtin_amdgcn_wmma_f32_16x16x32_f16(false, (A), false, (B), (short)0, (C), false, false)

// ---- CDNA5 async memory->LDS copy (ASYNCcnt path, ISA ch.10 / 08_async) ----
__device__ __forceinline__ void async_ld16(const char* g, unsigned lds_byte) {
    // per-lane: LDS[lds_byte] = MEM[g], 16 bytes, no VGPR data round-trip
    asm volatile("global_load_async_to_lds_b128 %0, %1, off"
                 :: "v"(lds_byte), "v"(g)
                 : "memory");
}
__device__ __forceinline__ void wait_async0() {
    asm volatile("s_wait_asynccnt 0x0" ::: "memory");
}

// --------------------------- f32 -> f16 convert ----------------------------
__global__ void k_cvt_f16(const float* __restrict__ src,
                          _Float16* __restrict__ dst, int n) {
    int i = blockIdx.x * blockDim.x + threadIdx.x;
    if (i < n) dst[i] = (_Float16)src[i];
}

// ----------------- W [512][1024] f32 -> Wt [1024][512] f16 -----------------
__global__ void k_transW(const float* __restrict__ W, _Float16* __restrict__ Wt) {
    int i = blockIdx.x * blockDim.x + threadIdx.x;   // over 1024*512
    int c = i >> 9;          // output row = W column
    int k = i & 511;         // output col = W row
    Wt[i] = (_Float16)W[(size_t)k * HD + c];
}

// ---------------------------------------------------------------------------
// Projection GEMM: C[16x16] = Xh[4096x512] @ Wt^T.  One wave per tile, K=512.
// outQK != null : head-major f16 [8][4096][128]  (q or k)
// outVT != null : transposed f16 vT[8][16][128][256]
// ---------------------------------------------------------------------------
__global__ void k_proj(const _Float16* __restrict__ Xh,   // [4096][512]
                       const _Float16* __restrict__ Wt,   // [1024][512]
                       _Float16* __restrict__ outQK,
                       _Float16* __restrict__ outVT) {
    const int lane  = threadIdx.x;        // 0..31 (wave32)
    const int mt    = blockIdx.x;         // 0..255  row tile
    const int ct    = blockIdx.y;         // 0..63   col tile
    const int lrow  = lane & 15;
    const int lhalf = lane >> 4;

    const _Float16* arow = Xh + (size_t)(mt * 16 + lrow) * D_IN;
    const _Float16* brow = Wt + (size_t)(ct * 16 + lrow) * D_IN;

    v8f acc = {0.f, 0.f, 0.f, 0.f, 0.f, 0.f, 0.f, 0.f};
    for (int kk = 0; kk < 16; ++kk) {
        const int ka = kk * 32 + lhalf * 16;
        v8h  a0 = *(const v8h*)(arow + ka);
        v8h  a1 = *(const v8h*)(arow + ka + 8);
        v16h af = SHUF16(a0, a1);
        v16h bf = *(const v16h*)(brow + kk * 32 + lhalf * 16);
        acc = WMMA_F16(af, bf, acc);
    }

    if (outQK) {                                           // head-major q/k
#pragma unroll
        for (int j = 0; j < 8; ++j) {
            const int m = mt * 16 + j + 8 * lhalf;
            const int c = ct * 16 + lrow;
            const int h = c >> 7, d = c & 127;
            outQK[((size_t)h * N_ROWS + m) * DH + d] = (_Float16)acc[j];
        }
    } else {                                               // transposed v
#pragma unroll
        for (int j = 0; j < 8; ++j) {
            const int m  = mt * 16 + j + 8 * lhalf;
            const int c  = ct * 16 + lrow;
            const int h  = c >> 7, d = c & 127;
            const int st = m >> 8, it = m & 255;
            outVT[(((size_t)(h * NSET + st) * DH) + d) * NITEM + it] = (_Float16)acc[j];
        }
    }
}

// ---------------------------------------------------------------------------
// Fused attention. Block = 8 waves = 8 q-row tiles of one (128-row group, set).
// Per head: async-stage K tile [256][128] + V^T tile [128][256] (128 KB) into
// LDS once, shared by all 8 waves. Each wave:
//   scores(16x256, K=128 WMMA) -> *conv_w[h]/(sqrt(Dh)*nItem) -> f16 via
//   per-wave LDS transpose -> acc[dt] += scores @ vT (K=256 WMMA).
// Head mixing lives in the folded scale; acc persists across heads.
// ---------------------------------------------------------------------------
__global__ void __launch_bounds__(256)
k_attn(const _Float16* __restrict__ Qhm,  // [8][4096][128]
       const _Float16* __restrict__ Khm,  // [8][4096][128]
       const _Float16* __restrict__ VT,   // [8][16][128][256]
       const float*    __restrict__ convw,// [8]
       float*          __restrict__ out)  // [4096][16][128]
{
    __shared__ __align__(32) _Float16 kbuf[NITEM * DH];        // 64 KB [item][d]
    __shared__ __align__(32) _Float16 vbuf[DH * NITEM];        // 64 KB [d][item]
    __shared__ __align__(32) _Float16 sScore[MTW][16 * 32];    // 8 KB, per-wave

    const int tid   = threadIdx.x;        // 0..255
    const int lane  = tid & 31;
    const int wid   = tid >> 5;           // 0..7
    const int mt    = blockIdx.x * MTW + wid;
    const int s     = blockIdx.y;         // 0..15
    const int lrow  = lane & 15;
    const int lhalf = lane >> 4;

    const unsigned kbuf_lds = (unsigned)(uintptr_t)&kbuf[0];
    const unsigned vbuf_lds = (unsigned)(uintptr_t)&vbuf[0];

    v8f acc[8];
#pragma unroll
    for (int dt = 0; dt < 8; ++dt)
        acc[dt] = (v8f){0.f, 0.f, 0.f, 0.f, 0.f, 0.f, 0.f, 0.f};

    const float scl_const = 0.088388347648318447f / 256.0f;  // 1/sqrt(128)/nItem

    for (int h = 0; h < NHEAD; ++h) {
        __syncthreads();   // previous iteration's LDS reads are done
        {
            // 64 KB contiguous K tile + 64 KB contiguous V^T tile -> LDS
            const char* kg = (const char*)(Khm + ((size_t)h * N_ROWS + s * NITEM) * DH);
            const char* vg = (const char*)(VT + (size_t)(h * NSET + s) * DH * NITEM);
#pragma unroll
            for (int i = 0; i < 16; ++i) {
                const unsigned off = (unsigned)(i * 4096 + tid * 16);
                async_ld16(kg + off, kbuf_lds + off);
                async_ld16(vg + off, vbuf_lds + off);
            }
            wait_async0();
        }
        __syncthreads();   // staged tiles visible to all waves

        const float hscale = convw[h] * scl_const;

        // q fragments for this head: 16 rows x K=128 -> 4 A-frags in regs
        const _Float16* qrow = Qhm + ((size_t)h * N_ROWS + mt * 16 + lrow) * DH;
        v16h aq[4];
#pragma unroll
        for (int kk = 0; kk < 4; ++kk) {
            const int ka = kk * 32 + lhalf * 16;
            v8h a0 = *(const v8h*)(qrow + ka);
            v8h a1 = *(const v8h*)(qrow + ka + 8);
            aq[kk] = SHUF16(a0, a1);
        }

        for (int kb = 0; kb < 8; ++kb) {                  // 32 items per step
            // ---- scores: two 16x16 tiles -> scaled f16 -> sScore[wid] ----
#pragma unroll
            for (int sub = 0; sub < 2; ++sub) {
                const int litem0 = kb * 32 + sub * 16;
                const _Float16* krow = &kbuf[(litem0 + lrow) * DH];
                v8f cs = {0.f, 0.f, 0.f, 0.f, 0.f, 0.f, 0.f, 0.f};
#pragma unroll
                for (int kk = 0; kk < 4; ++kk) {
                    v16h bk = *(const v16h*)(krow + kk * 32 + lhalf * 16);
                    cs = WMMA_F16(aq[kk], bk, cs);        // C[M=q row][N=item]
                }
#pragma unroll
                for (int j = 0; j < 8; ++j) {
                    const int Mr = j + 8 * lhalf;
                    sScore[wid][Mr * 32 + sub * 16 + lrow] = (_Float16)(cs[j] * hscale);
                }
            }
            // same-wave LDS ops are in-order: read the transposed chunk back
            v8h  s0 = *(const v8h*)(&sScore[wid][lrow * 32 + lhalf * 16]);
            v8h  s1 = *(const v8h*)(&sScore[wid][lrow * 32 + lhalf * 16 + 8]);
            v16h af = SHUF16(s0, s1);                     // A[M=q row][K=item]

            // ---- acc[dt] += scores_chunk @ v_chunk over all Dh tiles ----
#pragma unroll
            for (int dt = 0; dt < 8; ++dt) {
                const _Float16* vrow = &vbuf[(dt * 16 + lrow) * NITEM + kb * 32 + lhalf * 16];
                v16h bv = *(const v16h*)(vrow);           // B col = vT row (contig)
                acc[dt] = WMMA_F16(af, bv, acc[dt]);
            }
        }
    }

    // out[n][s][d], n = mt*16 + M, d = dt*16 + N
#pragma unroll
    for (int dt = 0; dt < 8; ++dt) {
#pragma unroll
        for (int j = 0; j < 8; ++j) {
            const int n = mt * 16 + j + 8 * lhalf;
            const int d = dt * 16 + lrow;
            out[((size_t)n * NSET + s) * DH + d] = acc[dt][j];
        }
    }
}

// ---------------------------------------------------------------------------
extern "C" void kernel_launch(void* const* d_in, const int* in_sizes, int n_in,
                              void* d_out, int out_size, void* d_ws, size_t ws_size,
                              hipStream_t stream) {
    (void)in_sizes; (void)n_in; (void)out_size; (void)ws_size;

    const float* x     = (const float*)d_in[0];
    const float* y     = (const float*)d_in[1];
    const float* W1    = (const float*)d_in[2];
    const float* W2    = (const float*)d_in[3];
    const float* W3    = (const float*)d_in[4];
    const float* convw = (const float*)d_in[5];
    float*       out   = (float*)d_out;

    char* ws = (char*)d_ws;
    const size_t MB = 1024ull * 1024ull;
    _Float16* xh  = (_Float16*)(ws + 0 * MB);   // 4 MB  [4096][512]
    _Float16* yh  = (_Float16*)(ws + 4 * MB);   // 4 MB
    _Float16* W1t = (_Float16*)(ws + 8 * MB);   // 1 MB  [1024][512]
    _Float16* W2t = (_Float16*)(ws + 9 * MB);   // 1 MB
    _Float16* W3t = (_Float16*)(ws + 10 * MB);  // 1 MB
    _Float16* qhm = (_Float16*)(ws + 11 * MB);  // 8 MB  [8][4096][128]
    _Float16* khm = (_Float16*)(ws + 19 * MB);  // 8 MB
    _Float16* vt  = (_Float16*)(ws + 27 * MB);  // 8 MB  [8][16][128][256]

    const int nxy = N_ROWS * D_IN;
    k_cvt_f16<<<(nxy + 255) / 256, 256, 0, stream>>>(x, xh, nxy);
    k_cvt_f16<<<(nxy + 255) / 256, 256, 0, stream>>>(y, yh, nxy);

    const int nw = D_IN * HD;
    k_transW<<<(nw + 255) / 256, 256, 0, stream>>>(W1, W1t);
    k_transW<<<(nw + 255) / 256, 256, 0, stream>>>(W2, W2t);
    k_transW<<<(nw + 255) / 256, 256, 0, stream>>>(W3, W3t);

    dim3 pg(N_ROWS / 16, HD / 16);              // 256 x 64 waves
    k_proj<<<pg, 32, 0, stream>>>(xh, W1t, qhm, nullptr);
    k_proj<<<pg, 32, 0, stream>>>(yh, W2t, khm, nullptr);
    k_proj<<<pg, 32, 0, stream>>>(yh, W3t, nullptr, vt);

    dim3 ag(N_ROWS / (16 * MTW), NSET);         // 32 x 16 blocks, 8 waves each
    k_attn<<<ag, 256, 0, stream>>>(qhm, khm, vt, convw, out);
}
